// SimpleRNN_67078799229447
// MI455X (gfx1250) — matrix operand order repaired and checked by reading, not compiled, not run on hardware
//
#include <hip/hip_runtime.h>
#include <hip/hip_bf16.h>

// ---------------------------------------------------------------------------
// Elman RNN scan on MI455X (gfx1250).
// 4 persistent workgroups (16 batch rows each). All weight WMMA B-fragments
// register-resident (32 frags = 256 VGPRs/wave, converted from global f32
// once). One barrier per step; per step: 24 ds_load_b128 (A-fragments, issued
// as one clause) feeding 32 v_wmma_f32_16x16x32_bf16 in 3 independent chains.
// The output head y(t-1) reuses the SAME h A-fragments as step t's Whh GEMM,
// so it costs zero extra LDS traffic (delayed-by-one-step emission).
// ---------------------------------------------------------------------------

#define B_  64
#define T_  2048
#define I_  128
#define H_  256
#define O_  128

typedef __bf16 bf16_t;
typedef __attribute__((ext_vector_type(16))) __bf16 v16bf;
typedef __attribute__((ext_vector_type(8)))  __bf16 v8bf;
typedef __attribute__((ext_vector_type(8)))  float  v8f;
typedef __attribute__((ext_vector_type(4)))  float  v4f;

// ---- WMMA fragment helpers (contiguous k order, k = 16*(lane>>4) + s, in A and B) -----------------

// B-matrix 32x16 (KxN) fragment straight from global f32 weights stored
// [N][K] row-major: lane holds column n = n0 + (lane&15), 16 contiguous K
// values starting at k0 + (lane>>4)*16. One-time load, converted to bf16.
__device__ __forceinline__ v16bf load_bfrag_global(const float* __restrict__ W,
                                                   int K, int n0, int k0,
                                                   int laneN, int laneHalf) {
  const float* p = W + (n0 + laneN) * K + k0 + laneHalf * 16;
  v16bf r;
#pragma unroll
  for (int i = 0; i < 16; ++i) r[i] = (bf16_t)p[i];
  return r;
}

// A-matrix 16x32 (MxK) fragment from LDS activations stored [16][K] row-major:
// lane: m = lane&15; K = {k0 + h*16 .. +15}, h=lane>>4.
__device__ __forceinline__ v16bf load_afrag(const bf16_t* Xl, int K, int k0,
                                            int laneN, int laneHalf) {
  const bf16_t* p = Xl + laneN * K + k0 + laneHalf * 16;
  v8bf lo = *(const v8bf*)p;
  v8bf hi = *(const v8bf*)(p + 8);
  return __builtin_shufflevector(lo, hi, 0, 1, 2, 3, 4, 5, 6, 7,
                                 8, 9, 10, 11, 12, 13, 14, 15);
}

__device__ __forceinline__ v8f wmma_bf16(v16bf a, v16bf b, v8f c) {
  return __builtin_amdgcn_wmma_f32_16x16x32_bf16(
      false, a, false, b, (short)0, c, false, false);
}

__device__ __forceinline__ v8f splat8(float v) {
  v8f r = {v, v, v, v, v, v, v, v};
  return r;
}

// ---------------------------------------------------------------------------

__global__ __launch_bounds__(256) void
SimpleRNN_67078799229447_kernel(const float* __restrict__ x,
                                const float* __restrict__ h0,
                                const float* __restrict__ Wxh,
                                const float* __restrict__ Whh,
                                const float* __restrict__ bh,
                                const float* __restrict__ Wout,
                                const float* __restrict__ bout,
                                float* __restrict__ out) {
  __shared__ bf16_t sX[2][16 * I_];   // 8 KB  ping-pong x tile (bf16)
  __shared__ bf16_t sH[2][16 * H_];   // 16 KB ping-pong hidden state (bf16)

  const int tid      = threadIdx.x;
  const int wave     = tid >> 5;
  const int lane     = tid & 31;
  const int laneN    = lane & 15;
  const int laneHalf = lane >> 4;
  const int m0       = blockIdx.x * 16;   // batch-row base of this chunk

  // Per-wave tile ownership: hidden N-tiles {2w, 2w+1}; output O-tile w.
  const int n0 = (wave * 2)     * 16;
  const int n1 = (wave * 2 + 1) * 16;
  const int o0 = wave * 16;

  // ---- one-time: weight fragments -> registers (32 frags, 256 VGPRs) ------
  v16bf fWxh0[4], fWxh1[4], fWhh0[8], fWhh1[8], fWout[8];
#pragma unroll
  for (int kk = 0; kk < 4; ++kk) {
    fWxh0[kk] = load_bfrag_global(Wxh, I_, n0, kk * 32, laneN, laneHalf);
    fWxh1[kk] = load_bfrag_global(Wxh, I_, n1, kk * 32, laneN, laneHalf);
  }
#pragma unroll
  for (int kk = 0; kk < 8; ++kk) {
    fWhh0[kk] = load_bfrag_global(Whh, H_, n0, kk * 32, laneN, laneHalf);
    fWhh1[kk] = load_bfrag_global(Whh, H_, n1, kk * 32, laneN, laneHalf);
    fWout[kk] = load_bfrag_global(Wout, H_, o0, kk * 32, laneN, laneHalf);
  }
  // Loop-invariant biases (one element per lane, per owned column).
  const float biasN0 = bh[n0 + laneN];
  const float biasN1 = bh[n1 + laneN];
  const float biasO  = bout[o0 + laneN];

  // ---- one-time: h0 and x[t=0] into LDS buffer 0 --------------------------
  for (int i = tid; i < 16 * H_; i += 256) {
    int m = i / H_, n = i % H_;
    sH[0][i] = (bf16_t)h0[(m0 + m) * H_ + n];
  }
  const int xrow = tid >> 4;            // 0..15
  const int xc8  = (tid & 15) * 8;      // 0..120
  {
    const v4f* p = (const v4f*)(x + ((size_t)(m0 + xrow) * T_ + 0) * I_ + xc8);
    v4f a = p[0], b = p[1];
    v8bf v;
    v[0] = (bf16_t)a[0]; v[1] = (bf16_t)a[1];
    v[2] = (bf16_t)a[2]; v[3] = (bf16_t)a[3];
    v[4] = (bf16_t)b[0]; v[5] = (bf16_t)b[1];
    v[6] = (bf16_t)b[2]; v[7] = (bf16_t)b[3];
    *(v8bf*)(&sX[0][xrow * I_ + xc8]) = v;
  }
  __syncthreads();

  float* Y  = out;
  float* HT = out + (size_t)B_ * T_ * O_;

#pragma unroll 1
  for (int t = 0; t < T_; ++t) {
    const int cur = t & 1, nxt = cur ^ 1;
    const bf16_t* xcur = sX[cur];
    bf16_t*       xnxt = sX[nxt];
    const bf16_t* hcur = sH[cur];   // state entering step t (= tanh of t-1)
    bf16_t*       hnxt = sH[nxt];

    // Prefetch x[t+1] (index clamped at the tail; last deposit is dead).
    const int tn = (t + 1 < T_) ? (t + 1) : (T_ - 1);
    const v4f* px =
        (const v4f*)(x + ((size_t)(m0 + xrow) * T_ + tn) * I_ + xc8);
    v4f pxa = px[0];
    v4f pxb = px[1];

    // ---- stage all A-fragments for this step as one LDS clause ------------
    v16bf ax[4], ah[8];
#pragma unroll
    for (int kk = 0; kk < 4; ++kk)
      ax[kk] = load_afrag(xcur, I_, kk * 32, laneN, laneHalf);
#pragma unroll
    for (int kk = 0; kk < 8; ++kk)
      ah[kk] = load_afrag(hcur, H_, kk * 32, laneN, laneHalf);

    // ---- output head y(t-1) = hcur @ Wout^T + bout: reuses ah, 0 LDS ------
    v8f yacc = splat8(biasO);
#pragma unroll
    for (int kk = 0; kk < 8; ++kk)
      yacc = wmma_bf16(ah[kk], fWout[kk], yacc);

    // ---- pre-activation a = x@Wxh^T + h@Whh^T + bh (two 16x16 N-tiles) ----
    v8f acc0 = splat8(biasN0);
    v8f acc1 = splat8(biasN1);
#pragma unroll
    for (int kk = 0; kk < 4; ++kk) {
      acc0 = wmma_bf16(ax[kk], fWxh0[kk], acc0);
      acc1 = wmma_bf16(ax[kk], fWxh1[kk], acc1);
    }
#pragma unroll
    for (int kk = 0; kk < 8; ++kk) {
      acc0 = wmma_bf16(ah[kk], fWhh0[kk], acc0);
      acc1 = wmma_bf16(ah[kk], fWhh1[kk], acc1);
    }

    // ---- emit y(t-1) (delayed one step; nothing to emit at t==0) ----------
    if (t > 0) {
#pragma unroll
      for (int r = 0; r < 8; ++r) {
        const int m = laneHalf * 8 + r;
        Y[((size_t)(m0 + m) * T_ + (t - 1)) * O_ + o0 + laneN] = yacc[r];
      }
    }

    // ---- h_new = tanh(a): C/D layout -> (m = laneHalf*8 + r, n = laneN) ---
#pragma unroll
    for (int r = 0; r < 8; ++r) {
      const int m = laneHalf * 8 + r;
      float hv0 = tanhf(acc0[r]);
      float hv1 = tanhf(acc1[r]);
      hnxt[m * H_ + n0 + laneN] = (bf16_t)hv0;
      hnxt[m * H_ + n1 + laneN] = (bf16_t)hv1;
      if (t == T_ - 1) {                 // final hidden state, full f32
        HT[(size_t)(m0 + m) * H_ + n0 + laneN] = hv0;
        HT[(size_t)(m0 + m) * H_ + n1 + laneN] = hv1;
      }
    }

    // Deposit prefetched x[t+1] into the nxt x-buffer before the barrier.
    {
      v8bf v;
      v[0] = (bf16_t)pxa[0]; v[1] = (bf16_t)pxa[1];
      v[2] = (bf16_t)pxa[2]; v[3] = (bf16_t)pxa[3];
      v[4] = (bf16_t)pxb[0]; v[5] = (bf16_t)pxb[1];
      v[6] = (bf16_t)pxb[2]; v[7] = (bf16_t)pxb[3];
      *(v8bf*)(&xnxt[xrow * I_ + xc8]) = v;
    }

    __syncthreads();  // sole barrier of the step
  }

  // ---- tail: y(T-1) from the final state (in sH[0], since T_ is even) -----
  {
    const bf16_t* hfin = sH[T_ & 1];
    v8f yacc = splat8(biasO);
#pragma unroll
    for (int kk = 0; kk < 8; ++kk) {
      v16bf a = load_afrag(hfin, H_, kk * 32, laneN, laneHalf);
      yacc = wmma_bf16(a, fWout[kk], yacc);
    }
#pragma unroll
    for (int r = 0; r < 8; ++r) {
      const int m = laneHalf * 8 + r;
      Y[((size_t)(m0 + m) * T_ + (T_ - 1)) * O_ + o0 + laneN] = yacc[r];
    }
  }
}

extern "C" void kernel_launch(void* const* d_in, const int* in_sizes, int n_in,
                              void* d_out, int out_size, void* d_ws,
                              size_t ws_size, hipStream_t stream) {
  (void)in_sizes; (void)n_in; (void)out_size; (void)d_ws; (void)ws_size;
  const float* x    = (const float*)d_in[0];
  const float* h0   = (const float*)d_in[1];
  const float* Wxh  = (const float*)d_in[2];
  const float* Whh  = (const float*)d_in[3];
  const float* bh   = (const float*)d_in[4];
  const float* Wout = (const float*)d_in[5];
  const float* bout = (const float*)d_in[6];
  float* out = (float*)d_out;

  SimpleRNN_67078799229447_kernel<<<B_ / 16, 256, 0, stream>>>(
      x, h0, Wxh, Whh, bh, Wout, bout, out);
}
